// CombinedLoss_19490561589353
// MI455X (gfx1250) — matrix operand order repaired and checked by reading, not compiled, not run on hardware
//
#include <hip/hip_runtime.h>
#include <hip/hip_bf16.h>
#include <math.h>

typedef __attribute__((ext_vector_type(16))) _Float16 v16h;
typedef __attribute__((ext_vector_type(8)))  _Float16 v8h;
typedef __attribute__((ext_vector_type(8)))  float    v8f;

// ---------------- scalar accumulator slots in workspace ----------------
enum {
  S_L1 = 0, S_MSE = 1, S_TSUM0 = 2, S_TSUM1 = 3, S_PSUM0 = 4, S_PSUM1 = 5,
  S_PERC = 6, S_H1 = 7, S_H2 = 8, S_HDIFF = 9, S_EXP = 10, S_HV = 11,
  S_WV = 12, S_SPAT = 13, S_SSIM_BASE = 16, /* 5 scales * 6 (b,c) * 2 */ S_COUNT = 96
};

// normalized 11-tap gaussian (sigma = 1.5)
__constant__ float GW[11] = {
  0.00102838f, 0.00759876f, 0.03600077f, 0.10936069f, 0.21300553f,
  0.26601172f, 0.21300553f, 0.10936069f, 0.03600077f, 0.00759876f, 0.00102838f
};

__device__ inline float waveReduce(float v) {
  for (int o = 16; o > 0; o >>= 1) v += __shfl_down(v, o);
  return v;
}
__device__ inline void atomicAddWave(float* p, float v) {
  v = waveReduce(v);
  if ((threadIdx.x & 31) == 0) atomicAdd(p, v);
}

__global__ void zero_kernel(float* p, int n) {
  int i = blockIdx.x * 256 + threadIdx.x;
  if (i < n) p[i] = 0.0f;
}

// ---------------- weight transpose + f16 convert -------------------------------
// in : fp32 [Cout][Cin][3][3]   out: f16 [Cout][tap(9)][Cin]
__global__ void wtrans_kernel(const float* __restrict__ w, _Float16* __restrict__ out,
                              int Cin, int Cout) {
  int nel = Cout * Cin * 9;
  int i = blockIdx.x * 256 + threadIdx.x;
  if (i >= nel) return;
  int tap = i % 9;
  int t2  = i / 9;
  int ci  = t2 % Cin;
  int cout = t2 / Cin;
  out[((size_t)cout * 9 + tap) * Cin + ci] = (_Float16)w[i];
}

// ---------------- WMMA implicit-GEMM 3x3 conv (SAME pad) + bias + ReLU ---------
// grid: (H*W/16, Cout/(nWaves*16), B)   block: THREADS (128 for Cout==64, else 256)
// Each wave: one 16(cout) x 16(spatial) f32 tile.
// K-loop is tap-major: 9 taps x ceil(Cin/32) chunks of WMMA 16x16x32.
// All 16 positions of a tile share one image row (W%16==0) -> row logic scalar.
// Full channel chunks are guard-free; column edges handled by clamp+select.
template <int THREADS>
__global__ void __launch_bounds__(THREADS)
conv3x3_wmma(const float* __restrict__ x, const _Float16* __restrict__ wt,
             const float* __restrict__ bias, float* __restrict__ y,
             int Cin, int Cout, int H, int W)
{
  constexpr int NWAVES = THREADS / 32;
  constexpr int NITER  = 512 / THREADS;   // staging elements per thread

  const int HW = H * W;
  const int b = blockIdx.z;
  x += (size_t)b * Cin * HW;
  y += (size_t)b * Cout * HW;

  const int nBase = blockIdx.x * 16;
  const int row0  = nBase / W;          // block-uniform
  const int col0  = nBase % W;

  const int lane   = threadIdx.x & 31;
  const int waveId = threadIdx.x >> 5;
  const int nTilesC  = Cout >> 4;
  const int coutBase = ((blockIdx.y * NWAVES + waveId) % nTilesC) * 16;

  const int n  = lane & 15;             // C/D column (spatial), A row M
  const int hi = lane >> 4;             // lane-half select
  const int kA = hi * 8;

  __shared__ __align__(32) _Float16 ldsB[16 * 32];   // [n][kk]

  const int coutRow = coutBase + n;
  const int cbFull  = Cin & ~31;

  v8f acc = {};

  #pragma unroll
  for (int t = 0; t < 9; ++t) {
    const int ky = t / 3 - 1;           // constants after unroll
    const int kx = t % 3 - 1;
    const int iy = row0 + ky;
    if (iy < 0 || iy >= H) continue;    // block-uniform scalar branch
    const float* xrow = x + (size_t)iy * W;
    const _Float16* wrow_t = wt + ((size_t)coutRow * 9 + t) * Cin;

    // ---- full 32-channel chunks: no channel guard, column clamp+select ----
    for (int cb = 0; cb < cbFull; cb += 32) {
      __syncthreads();
      #pragma unroll
      for (int it = 0; it < NITER; ++it) {
        int e  = threadIdx.x + it * THREADS;
        int kk = e >> 4, nn = e & 15;
        int ix  = col0 + nn + kx;
        int ixc = min(max(ix, 0), W - 1);
        float v = xrow[(size_t)(cb + kk) * HW + ixc];
        if ((unsigned)ix >= (unsigned)W) v = 0.0f;
        ldsB[nn * 32 + ((kk >> 3) & 1) * 16 + ((kk >> 4) << 3) + (kk & 7)] = (_Float16)v;
      }
      __syncthreads();

      const _Float16* wb = wrow_t + cb + kA;
      v8h lo = *(const v8h*)wb;
      v8h hh = *(const v8h*)(wb + 16);
      v16h a = __builtin_shufflevector(lo, hh, 0, 1, 2, 3, 4, 5, 6, 7,
                                               8, 9, 10, 11, 12, 13, 14, 15);
      v16h bf = *(const v16h*)&ldsB[n * 32 + hi * 16];
      acc = __builtin_amdgcn_wmma_f32_16x16x32_f16(false, a, false, bf,
                                                   (short)0, acc, false, false);
    }

    // ---- tail chunk (only Cin%32 != 0, i.e. conv0): clamp+select guards ----
    if (cbFull < Cin) {
      const int cb = cbFull;
      __syncthreads();
      #pragma unroll
      for (int it = 0; it < NITER; ++it) {
        int e  = threadIdx.x + it * THREADS;
        int kk = e >> 4, nn = e & 15;
        int ci  = cb + kk;
        int cic = min(ci, Cin - 1);
        int ix  = col0 + nn + kx;
        int ixc = min(max(ix, 0), W - 1);
        float v = xrow[(size_t)cic * HW + ixc];
        if (ci >= Cin || (unsigned)ix >= (unsigned)W) v = 0.0f;
        ldsB[nn * 32 + ((kk >> 3) & 1) * 16 + ((kk >> 4) << 3) + (kk & 7)] = (_Float16)v;
      }
      __syncthreads();

      v16h a;
      #pragma unroll
      for (int i = 0; i < 8; ++i) {
        int c0 = cb + kA + i, c1 = cb + kA + 16 + i;
        _Float16 w0 = wrow_t[min(c0, Cin - 1)];
        _Float16 w1 = wrow_t[min(c1, Cin - 1)];
        a[i]     = (c0 < Cin) ? w0 : (_Float16)0.0f;
        a[i + 8] = (c1 < Cin) ? w1 : (_Float16)0.0f;
      }
      v16h bf = *(const v16h*)&ldsB[n * 32 + hi * 16];
      acc = __builtin_amdgcn_wmma_f32_16x16x32_f16(false, a, false, bf,
                                                   (short)0, acc, false, false);
    }
  }

  // epilogue: bias + ReLU + store (VGPR r -> M = r + 8*hi)
  #pragma unroll
  for (int r = 0; r < 8; ++r) {
    int m = coutBase + r + hi * 8;
    float v = acc[r] + bias[m];
    y[(size_t)m * HW + (nBase + n)] = fmaxf(v, 0.0f);
  }
}

// ---------------- pooling kernels ----------------
__global__ void maxpool2(const float* __restrict__ x, float* __restrict__ y,
                         int planes, int H, int W) {
  int oH = H >> 1, oW = W >> 1;
  int N = planes * oH * oW;
  int i = blockIdx.x * 256 + threadIdx.x;
  if (i >= N) return;
  int ow = i % oW; int t = i / oW;
  int oh = t % oH; int pl = t / oH;
  const float* xp = x + ((size_t)pl * H + oh * 2) * W + ow * 2;
  y[i] = fmaxf(fmaxf(xp[0], xp[1]), fmaxf(xp[W], xp[W + 1]));
}

__global__ void avgpool2(const float* __restrict__ x, float* __restrict__ y,
                         int planes, int H, int W) {
  int oH = H >> 1, oW = W >> 1;
  int N = planes * oH * oW;
  int i = blockIdx.x * 256 + threadIdx.x;
  if (i >= N) return;
  int ow = i % oW; int t = i / oW;
  int oh = t % oH; int pl = t / oH;
  const float* xp = x + ((size_t)pl * H + oh * 2) * W + ow * 2;
  y[i] = 0.25f * (xp[0] + xp[1] + xp[W] + xp[W + 1]);
}

// ---------------- elementwise losses (smooth-L1, MSE, per-batch sums) ----------
__global__ void elem_losses(const float* __restrict__ yt, const float* __restrict__ yp,
                            int N, int perB, float* scal) {
  int i = blockIdx.x * 256 + threadIdx.x;
  float l1 = 0.f, mse = 0.f, ts = 0.f, ps = 0.f;
  int bsel = 0;
  if (i < N) {
    float t = yt[i], q = yp[i];
    float d = q - t, ad = fabsf(d);
    l1 = (ad < 1.0f) ? 0.5f * d * d : ad - 0.5f;
    mse = d * d;
    ts = t; ps = q;
    bsel = i / perB;            // perB % 32 == 0 -> wave-uniform
  }
  atomicAddWave(&scal[S_L1], l1);
  atomicAddWave(&scal[S_MSE], mse);
  float tsr = waveReduce(ts), psr = waveReduce(ps);
  if ((threadIdx.x & 31) == 0 && i < N) {
    atomicAdd(&scal[S_TSUM0 + bsel], tsr);
    atomicAdd(&scal[S_PSUM0 + bsel], psr);
  }
}

// ---------------- illumination smoothness (TV^2 sums on y_pred) ----------------
__global__ void illum_kernel(const float* __restrict__ yp, int planes, int H, int W,
                             float* scal) {
  int N = planes * H * W;
  int i = blockIdx.x * 256 + threadIdx.x;
  float hv = 0.f, wv = 0.f;
  if (i < N) {
    int wcol = i % W;
    int hrow = (i / W) % H;
    float v = yp[i];
    if (hrow < H - 1) { float d = yp[i + W] - v; hv = d * d; }
    if (wcol < W - 1) { float d = yp[i + 1] - v; wv = d * d; }
  }
  atomicAddWave(&scal[S_HV], hv);
  atomicAddWave(&scal[S_WV], wv);
}

// ---------------- soft histogram: pass 1 (totals), pass 2 (abs diff) ----------
__global__ void hist_pass1(const float* __restrict__ yt, const float* __restrict__ yp,
                           int N, float* scal) {
  int i = blockIdx.x * 256 + threadIdx.x;
  float s1 = 0.f, s2 = 0.f;
  if (i < N) {
    float a = yt[i], b = yp[i];
    for (int bin = 0; bin < 256; ++bin) {
      float e = bin * (1.0f / 255.0f);
      float da = (a - e) * 100.0f;
      float db = (b - e) * 100.0f;
      s1 += __expf(-0.5f * da * da);
      s2 += __expf(-0.5f * db * db);
    }
  }
  atomicAddWave(&scal[S_H1], s1);
  atomicAddWave(&scal[S_H2], s2);
}

__global__ void hist_pass2(const float* __restrict__ yt, const float* __restrict__ yp,
                           int M /* C*H*W */, const float* scal_in, float* scal) {
  int i = blockIdx.x * 256 + threadIdx.x;
  float acc = 0.f;
  if (i < M) {
    float invS1 = 1.0f / scal_in[S_H1];
    float invS2 = 1.0f / scal_in[S_H2];
    float t0 = yt[i], t1 = yt[M + i], p0 = yp[i], p1 = yp[M + i];
    for (int bin = 0; bin < 256; ++bin) {
      float e = bin * (1.0f / 255.0f);
      float a0 = (t0 - e) * 100.0f, a1 = (t1 - e) * 100.0f;
      float b0 = (p0 - e) * 100.0f, b1 = (p1 - e) * 100.0f;
      float h1 = __expf(-0.5f * a0 * a0) + __expf(-0.5f * a1 * a1);
      float h2 = __expf(-0.5f * b0 * b0) + __expf(-0.5f * b1 * b1);
      acc += fabsf(h1 * invS1 - h2 * invS2);
    }
  }
  atomicAddWave(&scal[S_HDIFF], acc);
}

// ---------------- exposure loss (16x16 pool of channel mean vs 0.6) ------------
__global__ void exposure_kernel(const float* __restrict__ yp, float* scal) {
  int i = blockIdx.x * 256 + threadIdx.x;   // 2*14*14 = 392
  float acc = 0.f;
  if (i < 392) {
    int b = i / 196, r = i % 196;
    int ph = r / 14, pw = r % 14;
    const float* base = yp + (size_t)b * 3 * 224 * 224;
    float s = 0.f;
    for (int c = 0; c < 3; ++c) {
      const float* cp = base + (size_t)c * 224 * 224 + (ph * 16) * 224 + pw * 16;
      for (int dy = 0; dy < 16; ++dy)
        for (int dx = 0; dx < 16; ++dx) s += cp[dy * 224 + dx];
    }
    float v = s * (1.0f / 768.0f) - 0.6f;
    acc = v * v;
  }
  atomicAddWave(&scal[S_EXP], acc);
}

// ---------------- spatial consistency: 4x4 pooled luminance + 4 diff kernels ---
__global__ void pool_lum4(const float* __restrict__ x, float* __restrict__ out) {
  int i = blockIdx.x * 256 + threadIdx.x;   // 2*56*56
  if (i >= 2 * 56 * 56) return;
  int b = i / 3136, r = i % 3136;
  int h = r / 56, wcol = r % 56;
  const float* base = x + (size_t)b * 3 * 224 * 224;
  float s = 0.f;
  for (int c = 0; c < 3; ++c) {
    const float* cp = base + (size_t)c * 224 * 224 + (h * 4) * 224 + wcol * 4;
    for (int dy = 0; dy < 4; ++dy)
      for (int dx = 0; dx < 4; ++dx) s += cp[dy * 224 + dx];
  }
  out[i] = s * (1.0f / 48.0f);
}

__global__ void spat_kernel(const float* __restrict__ op, const float* __restrict__ ep,
                            float* scal) {
  int i = blockIdx.x * 256 + threadIdx.x;   // 2*56*56
  float acc = 0.f;
  if (i < 2 * 56 * 56) {
    int b = i / 3136, r = i % 3136;
    int h = r / 56, wcol = r % 56;
    const float* o = op + b * 3136;
    const float* e = ep + b * 3136;
    float co = o[r], ce = e[r];
    #define NV(m, hh, ww) (((hh) >= 0 && (hh) < 56 && (ww) >= 0 && (ww) < 56) ? m[(hh) * 56 + (ww)] : 0.0f)
    float d;
    d = (co - NV(o, h, wcol - 1)) - (ce - NV(e, h, wcol - 1)); acc += d * d;
    d = (co - NV(o, h, wcol + 1)) - (ce - NV(e, h, wcol + 1)); acc += d * d;
    d = (co - NV(o, h - 1, wcol)) - (ce - NV(e, h - 1, wcol)); acc += d * d;
    d = (co - NV(o, h + 1, wcol)) - (ce - NV(e, h + 1, wcol)); acc += d * d;
    #undef NV
  }
  atomicAddWave(&scal[S_SPAT], acc);
}

// ---------------- SSIM: separable 11-tap gaussian, two passes ------------------
__global__ void ssim_vpass(const float* __restrict__ X, const float* __restrict__ Y,
                           int H, int W, float* __restrict__ vout) {
  int oH = H - 10;
  int tot = 6 * oH * W;
  int i = blockIdx.x * 256 + threadIdx.x;
  if (i >= tot) return;
  int wcol = i % W; int t = i / W;
  int hh = t % oH;  int bc = t / oH;
  const float* xp = X + (size_t)bc * H * W + hh * W + wcol;
  const float* yp = Y + (size_t)bc * H * W + hh * W + wcol;
  float sx = 0, sy = 0, sxx = 0, syy = 0, sxy = 0;
  #pragma unroll
  for (int k = 0; k < 11; ++k) {
    float g = GW[k];
    float xv = xp[k * W], yv = yp[k * W];
    sx += g * xv; sy += g * yv;
    sxx += g * xv * xv; syy += g * yv * yv; sxy += g * xv * yv;
  }
  size_t plane = (size_t)6 * oH * W;
  vout[i] = sx; vout[plane + i] = sy; vout[2 * plane + i] = sxx;
  vout[3 * plane + i] = syy; vout[4 * plane + i] = sxy;
}

__global__ void ssim_hpass(const float* __restrict__ vin, int oH, int W, int scale,
                           float* scal) {
  int oW = W - 10;
  int tot = 6 * oH * oW;
  int i = blockIdx.x * 256 + threadIdx.x;
  if (i >= tot) return;
  int wcol = i % oW; int t = i / oW;
  int hh = t % oH;   int bc = t / oH;
  size_t plane = (size_t)6 * oH * W;
  const float* p = vin + (size_t)bc * oH * W + hh * W + wcol;
  float mu1 = 0, mu2 = 0, e11 = 0, e22 = 0, e12 = 0;
  #pragma unroll
  for (int k = 0; k < 11; ++k) {
    float g = GW[k];
    mu1 += g * p[k];             mu2 += g * p[plane + k];
    e11 += g * p[2 * plane + k]; e22 += g * p[3 * plane + k];
    e12 += g * p[4 * plane + k];
  }
  float s11 = e11 - mu1 * mu1, s22 = e22 - mu2 * mu2, s12 = e12 - mu1 * mu2;
  const float C1 = 1e-4f, C2 = 9e-4f;
  float cs = (2.0f * s12 + C2) / (s11 + s22 + C2);
  float ss = ((2.0f * mu1 * mu2 + C1) / (mu1 * mu1 + mu2 * mu2 + C1)) * cs;
  atomicAdd(&scal[S_SSIM_BASE + (scale * 6 + bc) * 2 + 0], ss);
  atomicAdd(&scal[S_SSIM_BASE + (scale * 6 + bc) * 2 + 1], cs);
}

// ---------------- perceptual (VGG feature MSE) --------------------------------
__global__ void perc_kernel(const float* __restrict__ f1, const float* __restrict__ f2,
                            int N, float* scal) {
  int i = blockIdx.x * 256 + threadIdx.x;
  float v = 0.f;
  if (i < N) { float d = f1[i] - f2[i]; v = d * d; }
  atomicAddWave(&scal[S_PERC], v);
}

// ---------------- final scalar combine ----------------------------------------
__global__ void finalize_kernel(const float* scal, float* out) {
  if (threadIdx.x != 0 || blockIdx.x != 0) return;
  const float msw[5] = {0.0448f, 0.2856f, 0.3001f, 0.2363f, 0.1333f};
  const int   ssz[5] = {214, 102, 46, 18, 4};
  float ms_mean = 0.f;
  for (int bc = 0; bc < 6; ++bc) {
    float prod = 1.f;
    for (int s = 0; s < 5; ++s) {
      float cnt = (float)ssz[s] * (float)ssz[s];
      float sv = scal[S_SSIM_BASE + (s * 6 + bc) * 2 + 0] / cnt;
      float cv = scal[S_SSIM_BASE + (s * 6 + bc) * 2 + 1] / cnt;
      float v = (s == 4) ? sv : cv;
      v = v > 0.f ? v : 0.f;
      prod *= __powf(v, msw[s]);
    }
    ms_mean += prod;
  }
  ms_mean *= (1.0f / 6.0f);
  float msssim_l = 1.0f - ms_mean;

  const float Ntot = 2.0f * 3.0f * 224.0f * 224.0f;
  float l1     = scal[S_L1] / Ntot;
  float mse    = scal[S_MSE] / Ntot;
  float psnr_l = 40.0f + 10.0f * log10f(mse);
  float perc   = scal[S_PERC] / (2.0f * 256.0f * 56.0f * 56.0f);
  float hist_l = scal[S_HDIFF] / (3.0f * 224.0f * 224.0f * 256.0f);
  float perB   = 3.0f * 224.0f * 224.0f;
  float color_l = 0.5f * (fabsf(scal[S_TSUM0] - scal[S_PSUM0]) +
                          fabsf(scal[S_TSUM1] - scal[S_PSUM1])) / perB;
  float exp_l  = scal[S_EXP] / (2.0f * 14.0f * 14.0f);
  float ill    = 2.0f * (scal[S_HV] / (223.0f * 3.0f) + scal[S_WV] / (224.0f * 2.0f)) / 2.0f;
  float spat_m = scal[S_SPAT] / (2.0f * 56.0f * 56.0f);

  out[0] = l1 + 0.06f * perc + 0.05f * hist_l + 0.0083f * psnr_l + 0.25f * color_l
         + 0.5f * msssim_l + 0.1f * exp_l + 0.1f * ill + 0.1f * spat_m;
}

// =============================== launcher =====================================
extern "C" void kernel_launch(void* const* d_in, const int* in_sizes, int n_in,
                              void* d_out, int out_size, void* d_ws, size_t ws_size,
                              hipStream_t stream) {
  (void)in_sizes; (void)n_in; (void)out_size; (void)ws_size;
  const float* yt = (const float*)d_in[0];
  const float* yp = (const float*)d_in[1];
  float* out = (float*)d_out;

  float* ws = (float*)d_ws;
  const size_t FB = (size_t)64 * 224 * 224 * 2;   // 6,422,528 floats
  float* bufA = ws;
  float* bufB = bufA + FB;
  float* f1   = bufB + FB;                         // 2*256*56*56 = 1,605,632
  float* vbuf = f1 + 1605632;                      // 5 * 6*214*224 = 1,438,080
  float* Xa   = vbuf + 1440000;                    // 6*112*112 = 75,264 each
  float* Xb   = Xa + 75264;
  float* Ya   = Xb + 75264;
  float* Yb   = Ya + 75264;
  float* opb  = Yb + 75264;                        // 2*56*56 = 6272
  float* epb  = opb + 6272;
  float* scal = epb + 6272;                        // S_COUNT floats
  _Float16* wtBase = (_Float16*)(scal + S_COUNT);  // f16 transposed weights

  // per-layer transposed weight sizes (Cout*Cin*9), all multiples of 8
  const int wsz[7] = {1728, 36864, 73728, 147456, 294912, 589824, 589824};
  _Float16* wtL[7];
  { size_t off = 0;
    for (int l = 0; l < 7; ++l) { wtL[l] = wtBase + off; off += (size_t)wsz[l]; } }

  zero_kernel<<<1, 128, 0, stream>>>(scal, S_COUNT);

  // ---- transpose + f16-convert all VGG weights (cheap, deterministic) ----
  {
    const int cins[7]  = {3, 64, 64, 128, 128, 256, 256};
    const int couts[7] = {64, 64, 128, 128, 256, 256, 256};
    for (int l = 0; l < 7; ++l)
      wtrans_kernel<<<(wsz[l] + 255) / 256, 256, 0, stream>>>(
          (const float*)d_in[2 + 2 * l], wtL[l], cins[l], couts[l]);
  }

  // ---- elementwise / histogram / exposure / illumination ----
  const int N = 2 * 3 * 224 * 224;      // 301,056
  const int M = 3 * 224 * 224;          // 150,528
  elem_losses<<<N / 256, 256, 0, stream>>>(yt, yp, N, M, scal);
  illum_kernel<<<N / 256, 256, 0, stream>>>(yp, 6, 224, 224, scal);
  hist_pass1<<<N / 256, 256, 0, stream>>>(yt, yp, N, scal);
  hist_pass2<<<(M + 255) / 256, 256, 0, stream>>>(yt, yp, M, scal, scal);
  exposure_kernel<<<2, 256, 0, stream>>>(yp, scal);

  // ---- spatial consistency ----
  pool_lum4<<<(6272 + 255) / 256, 256, 0, stream>>>(yt, opb);
  pool_lum4<<<(6272 + 255) / 256, 256, 0, stream>>>(yp, epb);
  spat_kernel<<<(6272 + 255) / 256, 256, 0, stream>>>(opb, epb, scal);

  // ---- MS-SSIM: 5 scales ----
  {
    const float* Xc = yt; const float* Yc = yp;
    int dim = 224;
    for (int s = 0; s < 5; ++s) {
      int oH = dim - 10;
      int nv = 6 * oH * dim;
      ssim_vpass<<<(nv + 255) / 256, 256, 0, stream>>>(Xc, Yc, dim, dim, vbuf);
      int nh = 6 * oH * oH;
      ssim_hpass<<<(nh + 255) / 256, 256, 0, stream>>>(vbuf, oH, dim, s, scal);
      if (s < 4) {
        int nd = dim >> 1;
        float* xd = (s & 1) ? Xb : Xa;
        float* yd = (s & 1) ? Yb : Ya;
        int no = 6 * nd * nd;
        avgpool2<<<(no + 255) / 256, 256, 0, stream>>>(Xc, xd, 6, dim, dim);
        avgpool2<<<(no + 255) / 256, 256, 0, stream>>>(Yc, yd, 6, dim, dim);
        Xc = xd; Yc = yd; dim = nd;
      }
    }
  }

  // ---- VGG19 features[:16] via WMMA implicit GEMM, then perceptual MSE ----
  auto runconv = [&](const float* x, int li, int Cin, int Cout, int S, float* y) {
    const _Float16* wtp = wtL[li];
    const float* bi = (const float*)d_in[3 + 2 * li];
    if (Cout == 64) {   // 4 waves -> exact tile mapping
      dim3 g((unsigned)(S * S / 16), 1u, 2u);
      conv3x3_wmma<128><<<g, 128, 0, stream>>>(x, wtp, bi, y, Cin, Cout, S, S);
    } else {            // 8 waves
      dim3 g((unsigned)(S * S / 16), (unsigned)(Cout / 128), 2u);
      conv3x3_wmma<256><<<g, 256, 0, stream>>>(x, wtp, bi, y, Cin, Cout, S, S);
    }
  };
  for (int t = 0; t < 2; ++t) {
    const float* src = (t == 0) ? yt : yp;
    runconv(src,  0,   3,  64, 224, bufA);
    runconv(bufA, 1,  64,  64, 224, bufB);
    { int n = 2 * 64 * 112 * 112;
      maxpool2<<<(n + 255) / 256, 256, 0, stream>>>(bufB, bufA, 2 * 64, 224, 224); }
    runconv(bufA, 2,  64, 128, 112, bufB);
    runconv(bufB, 3, 128, 128, 112, bufA);
    { int n = 2 * 128 * 56 * 56;
      maxpool2<<<(n + 255) / 256, 256, 0, stream>>>(bufA, bufB, 2 * 128, 112, 112); }
    runconv(bufB, 4, 128, 256, 56, bufA);
    runconv(bufA, 5, 256, 256, 56, bufB);
    runconv(bufB, 6, 256, 256, 56, bufA);
    if (t == 0)
      hipMemcpyAsync(f1, bufA, (size_t)2 * 256 * 56 * 56 * sizeof(float),
                     hipMemcpyDeviceToDevice, stream);
  }
  { int n = 2 * 256 * 56 * 56;
    perc_kernel<<<(n + 255) / 256, 256, 0, stream>>>(f1, bufA, n, scal); }

  finalize_kernel<<<1, 32, 0, stream>>>(scal, out);
}
